// GAT_32916629356560
// MI455X (gfx1250) — matrix operand order repaired and checked by reading, not compiled, not run on hardware
//
#include <hip/hip_runtime.h>
#include <cstdint>
#include <cstddef>

// ---------------------------------------------------------------------------
// GAT 3-layer forward for MI455X (gfx1250).
//  - Dense lifts X@W via v_wmma_f32_16x16x32_bf16 (bf16 in, fp32 acc).
//  - B tile staged per-block into LDS with global_load_async_to_lds_b128
//    (ASYNCcnt path), padded pitch => bank-conflict-free ds_load_b128 frags.
//    The LDS object pointer is escaped into the asm so the compiler cannot
//    fold the fragment loads away (asm writes LDS invisibly otherwise).
//  - Edge softmax/aggregate via L2-resident float scatter atomics
//    (feature matrices are 25.6MB => fit the 192MB L2).
//  - Bias dropped (cancels inside BatchNorm); BN-apply fuses ReLU and the
//    f32->bf16 conversion feeding the next layer's WMMA GEMM.
// ---------------------------------------------------------------------------

typedef __bf16 bf16_t;
typedef bf16_t v16bf  __attribute__((ext_vector_type(16)));
typedef bf16_t bf16x8 __attribute__((ext_vector_type(8)));
typedef float  v8f    __attribute__((ext_vector_type(8)));

#define NEG_SLOPE 0.2f
#define BN_EPS 1e-5f

__device__ __forceinline__ unsigned short f32_to_bf16_bits(float x) {
  unsigned u = __float_as_uint(x);
  u += 0x7FFFu + ((u >> 16) & 1u);   // round-to-nearest-even
  return (unsigned short)(u >> 16);
}

__device__ __forceinline__ void atomicMaxF(float* addr, float v) {
  // IEEE ordered-int trick; valid with -inf initialization.
  if (v >= 0.0f) atomicMax((int*)addr, __float_as_int(v));
  else           atomicMin((unsigned int*)addr, __float_as_uint(v));
}

// ---------------- small utility kernels ----------------
__global__ void k_f32_to_bf16(const float* __restrict__ in,
                              unsigned short* __restrict__ out, int n) {
  int i = blockIdx.x * blockDim.x + threadIdx.x;
  if (i < n) out[i] = f32_to_bf16_bits(in[i]);
}

// W [K,F] fp32 row-major  ->  Wt [F,K] bf16 (column-contiguous for WMMA B frags)
__global__ void k_wt_bf16(const float* __restrict__ W,
                          unsigned short* __restrict__ Wt, int K, int F) {
  int i = blockIdx.x * blockDim.x + threadIdx.x;
  if (i < K * F) {
    int k = i / F, f = i - k * F;
    Wt[(size_t)f * K + k] = f32_to_bf16_bits(W[i]);
  }
}

__global__ void k_fill(float* __restrict__ p, float v, int n) {
  int i = blockIdx.x * blockDim.x + threadIdx.x;
  if (i < n) p[i] = v;
}

// ---------------- WMMA GEMM:  C[M,Nf] = A[M,K](bf16) * Bt[Nf,K](bf16) -------
// block = 256 threads = 8 waves; each wave -> 16 rows x 64 cols; grid.y tiles Nf.
// B tile (64 rows of Bt, contiguous 2*64*K bytes) is staged into LDS with the
// gfx1250 async-to-LDS path, with a 16B/row pad so fragment reads are
// bank-conflict-free (lane stride 2K+16 bytes => 4-bank stagger, 16 lanes
// cover all 64 banks).
#define SB_BYTES (64 * (256 * 2 + 16))
__global__ __launch_bounds__(256)
void k_gemm_bf16_wmma(const bf16_t* __restrict__ A, const bf16_t* __restrict__ Bt,
                      float* __restrict__ C, int M, int Nf, int K) {
  __shared__ char sB[SB_BYTES];              // max K=256: 33 KB (sole LDS object)

  const int tid   = threadIdx.x;
  const int lane  = tid & 31;
  const int wave  = tid >> 5;
  const int lid   = lane & 15;
  const int hsel  = lane >> 4;               // half-wave select (0/1)
  const int mBase = blockIdx.x * 128 + wave * 16;
  const int nBase = blockIdx.y * 64;
  const int pitch = 2 * K + 16;              // padded LDS row pitch (bytes)

  // --- async stage of B tile: 64*K bf16, 16B per lane per issue -------------
  {
    const unsigned long long bBase =
        (unsigned long long)(size_t)(Bt + (size_t)nBase * K);
    const int rowBytes = 2 * K;                    // power of two (256/512)
    const int rowShift = __popc(rowBytes - 1);     // log2(rowBytes)
    const int iters    = K >> 5;                   // (64*K*2) / (256*16)
    for (int it = 0; it < iters; ++it) {
      unsigned g      = (unsigned)(it * 4096 + tid * 16);  // linear src offset
      unsigned row    = g >> rowShift;
      unsigned within = g & (rowBytes - 1);
      unsigned l      = row * pitch + within;              // padded LDS offset
      // %3 (pointer to sB) is unused in the template but escapes the LDS
      // object into the asm, so the compiler must treat sB as written here.
      asm volatile("global_load_async_to_lds_b128 %0, %1, %2"
                   :: "v"(l), "v"(g), "s"(bBase), "v"((const void*)sB)
                   : "memory");
    }
    // visible store into the pad region: sB can never be proven store-free
    if (tid == 255) sB[SB_BYTES - 1] = 1;
    asm volatile("s_wait_asynccnt 0" ::: "memory");
    __syncthreads();
  }

  int mRow = mBase + lid;
  int mCl  = mRow < M ? mRow : M - 1;        // clamp loads; stores predicated

  v8f acc[4] = {};

  // A frag (16x32 bf16): lane holds row lid; K groups [hsel*8..+7], [16+hsel*8..+7]
  const bf16_t* aRow = A + (size_t)mCl * K + hsel * 8;

  for (int kc = 0; kc < K; kc += 32) {
    __builtin_prefetch(aRow + kc + 128, 0, 3);
    bf16x8 alo = *(const bf16x8*)(aRow + kc);
    bf16x8 ahi = *(const bf16x8*)(aRow + kc + 16);

    // B frags for all 4 N-tiles first (batched ds_load_b128, one wait)
    v16bf bvv[4];
#pragma unroll
    for (int nt = 0; nt < 4; ++nt) {
      const char* p = sB + (nt * 16 + lid) * pitch + (kc + hsel * 8) * 2;
      bf16x8 blo = *(const bf16x8*)(p);
      bf16x8 bhi = *(const bf16x8*)(p + 32);
#pragma unroll
      for (int i = 0; i < 8; ++i) { bvv[nt][i] = blo[i]; bvv[nt][8 + i] = bhi[i]; }
    }

    v16bf av;
#pragma unroll
    for (int i = 0; i < 8; ++i) { av[i] = alo[i]; av[8 + i] = ahi[i]; }

#pragma unroll
    for (int nt = 0; nt < 4; ++nt)
      acc[nt] = __builtin_amdgcn_wmma_f32_16x16x32_bf16(
          false, av, false, bvv[nt], (short)0, acc[nt], false, false);
  }

  // D layout: VGPR r -> m = r + 8*hsel (in tile), n = lid
#pragma unroll
  for (int nt = 0; nt < 4; ++nt) {
    int col = nBase + nt * 16 + lid;
#pragma unroll
    for (int r = 0; r < 8; ++r) {
      int m = mBase + r + 8 * hsel;
      if (m < M) C[(size_t)m * Nf + col] = acc[nt][r];
    }
  }
}

// ---------------- attention logits:  es/ed [N,H] ----------------
__global__ void k_logits(const float* __restrict__ h, const float* __restrict__ a_src,
                         const float* __restrict__ a_dst, float* __restrict__ es,
                         float* __restrict__ ed, int Nn, int F, int C,
                         int hShift, int Hmask) {
  int id = blockIdx.x * blockDim.x + threadIdx.x;
  if (id >= (Nn << hShift)) return;
  int n  = id >> hShift;
  int hh = id & Hmask;
  const float* row = h + (size_t)n * F + hh * C;
  const float* as  = a_src + hh * C;
  const float* ad  = a_dst + hh * C;
  float s1 = 0.f, s2 = 0.f;
  for (int c = 0; c < C; ++c) { float v = row[c]; s1 += v * as[c]; s2 += v * ad[c]; }
  es[id] = s1;
  ed[id] = s2;
}

// ---------------- segment-softmax passes over edges ----------------
__device__ __forceinline__ void edge_pair(const int* __restrict__ ei, int e, int E,
                                          int& src, int& dst) {
  if (e < E) { src = ei[e]; dst = ei[E + e]; }
  else       { src = dst = e - E; }           // self loops appended
}

__global__ void k_edge_max(const int* __restrict__ ei, const float* __restrict__ es,
                           const float* __restrict__ ed, float* __restrict__ m,
                           int E, int Etot, int hShift, int Hmask) {
  int id = blockIdx.x * blockDim.x + threadIdx.x;
  if (id >= (Etot << hShift)) return;
  int e = id >> hShift, hh = id & Hmask, H = Hmask + 1;
  int src, dst; edge_pair(ei, e, E, src, dst);
  float x = es[src * H + hh] + ed[dst * H + hh];
  x = fmaxf(x, NEG_SLOPE * x);                // leaky_relu (slope < 1)
  atomicMaxF(&m[dst * H + hh], x);
}

__global__ void k_edge_sum(const int* __restrict__ ei, const float* __restrict__ es,
                           const float* __restrict__ ed, const float* __restrict__ m,
                           float* __restrict__ s, int E, int Etot,
                           int hShift, int Hmask) {
  int id = blockIdx.x * blockDim.x + threadIdx.x;
  if (id >= (Etot << hShift)) return;
  int e = id >> hShift, hh = id & Hmask, H = Hmask + 1;
  int src, dst; edge_pair(ei, e, E, src, dst);
  float x = es[src * H + hh] + ed[dst * H + hh];
  x = fmaxf(x, NEG_SLOPE * x);
  atomicAdd(&s[dst * H + hh], __expf(x - m[dst * H + hh]));
}

// one wave per edge; lane covers F/32 contiguous channels (one head slice)
__global__ __launch_bounds__(256)
void k_edge_agg(const int* __restrict__ ei, const float* __restrict__ es,
                const float* __restrict__ ed, const float* __restrict__ m,
                const float* __restrict__ s, const float* __restrict__ h,
                float* __restrict__ agg, int E, int Etot, int F, int H,
                int cphShift) {
  int e = blockIdx.x * 8 + (threadIdx.x >> 5);
  if (e >= Etot) return;
  int lane = threadIdx.x & 31;
  int src, dst; edge_pair(ei, e, E, src, dst);
  int VEC  = F >> 5;
  int c0   = lane * VEC;
  int head = c0 >> cphShift;
  float x = es[src * H + head] + ed[dst * H + head];
  x = fmaxf(x, NEG_SLOPE * x);
  float alpha = __expf(x - m[dst * H + head]) / s[dst * H + head];
  const float* hs = h + (size_t)src * F + c0;
  float* ag = agg + (size_t)dst * F + c0;
  for (int j = 0; j < VEC; ++j) atomicAdd(ag + j, hs[j] * alpha);
}

// ---------------- BatchNorm (training-mode batch stats) ----------------
__global__ __launch_bounds__(256)
void k_bn_stats(const float* __restrict__ X, int Nn, int F,
                float* __restrict__ mu, float* __restrict__ rs) {
  __shared__ float s1[256], s2[256];
  int f = blockIdx.x;
  float a = 0.f, b = 0.f;
  for (int n = threadIdx.x; n < Nn; n += 256) {
    float v = X[(size_t)n * F + f];
    a += v; b += v * v;
  }
  s1[threadIdx.x] = a; s2[threadIdx.x] = b;
  __syncthreads();
  for (int off = 128; off > 0; off >>= 1) {
    if ((int)threadIdx.x < off) {
      s1[threadIdx.x] += s1[threadIdx.x + off];
      s2[threadIdx.x] += s2[threadIdx.x + off];
    }
    __syncthreads();
  }
  if (threadIdx.x == 0) {
    float mean = s1[0] / (float)Nn;
    float var  = s2[0] / (float)Nn - mean * mean;
    mu[f] = mean;
    rs[f] = rsqrtf(var + BN_EPS);
  }
}

__global__ void k_bn_apply(const float* __restrict__ X, const float* __restrict__ g,
                           const float* __restrict__ be, const float* __restrict__ mu,
                           const float* __restrict__ rs,
                           unsigned short* __restrict__ outBf,
                           float* __restrict__ outF, int total, int Fmask,
                           int doRelu) {
  int i = blockIdx.x * blockDim.x + threadIdx.x;
  if (i >= total) return;
  int f = i & Fmask;
  float y = g[f] * (X[i] - mu[f]) * rs[f] + be[f];
  if (doRelu) y = fmaxf(y, 0.f);
  if (outBf) outBf[i] = f32_to_bf16_bits(y);   // feeds next WMMA GEMM
  else       outF[i]  = y;                     // final output
}

// ---------------------------------------------------------------------------
extern "C" void kernel_launch(void* const* d_in, const int* in_sizes, int n_in,
                              void* d_out, int out_size, void* d_ws, size_t ws_size,
                              hipStream_t stream) {
  (void)n_in; (void)out_size; (void)ws_size;
  const int F_IN = 256, HID = 128;
  const int N    = in_sizes[0] / F_IN;
  const int E    = in_sizes[1] / 2;
  const int Etot = E + N;

  const float* x   = (const float*)d_in[0];
  const int*   ei  = (const int*)d_in[1];
  const float* W1  = (const float*)d_in[2];
  const float* as1 = (const float*)d_in[3];
  const float* ad1 = (const float*)d_in[4];
  // d_in[5] = b1  : dropped (cancels inside BN)
  const float* g1  = (const float*)d_in[6];
  const float* be1 = (const float*)d_in[7];
  const float* W2  = (const float*)d_in[8];
  const float* as2 = (const float*)d_in[9];
  const float* ad2 = (const float*)d_in[10];
  const float* g2  = (const float*)d_in[12];
  const float* be2 = (const float*)d_in[13];
  const float* W3  = (const float*)d_in[14];
  const float* as3 = (const float*)d_in[15];
  const float* ad3 = (const float*)d_in[16];
  const float* g3  = (const float*)d_in[18];
  const float* be3 = (const float*)d_in[19];

  size_t off = 0;
  auto alloc = [&](size_t bytes) -> void* {
    void* p = (char*)d_ws + off;
    off += (bytes + 255) & ~(size_t)255;
    return p;
  };
  unsigned short* xbf  = (unsigned short*)alloc((size_t)N * F_IN * 2); // GEMM input (bf16)
  float*          hbuf = (float*)alloc((size_t)N * HID * 4);           // h = X@W
  float*          agg  = (float*)alloc((size_t)N * HID * 4);           // aggregation out
  float*          es   = (float*)alloc((size_t)N * 4 * 4);
  float*          ed   = (float*)alloc((size_t)N * 4 * 4);
  float*          mbuf = (float*)alloc((size_t)N * 4 * 4);
  float*          sbuf = (float*)alloc((size_t)N * 4 * 4);
  unsigned short* wt   = (unsigned short*)alloc((size_t)F_IN * HID * 2);
  float*          mu   = (float*)alloc(HID * 4);
  float*          rs   = (float*)alloc(HID * 4);

  auto cdiv = [](long a, long b) { return (int)((a + b - 1) / b); };

  k_f32_to_bf16<<<cdiv((long)N * F_IN, 256), 256, 0, stream>>>(x, xbf, N * F_IN);

  struct Cfg { const float *W,*as,*ad,*g,*be; int K,F,H,C,hShift,cphShift,relu; };
  const Cfg L[3] = {
      {W1, as1, ad1, g1, be1, 256, 128, 4, 32, 2, 5, 1},
      {W2, as2, ad2, g2, be2, 128, 128, 4, 32, 2, 5, 1},
      {W3, as3, ad3, g3, be3, 128,  64, 1, 64, 0, 6, 0},
  };

  for (int li = 0; li < 3; ++li) {
    const Cfg& c = L[li];
    k_wt_bf16<<<cdiv((long)c.K * c.F, 256), 256, 0, stream>>>(c.W, wt, c.K, c.F);
    dim3 gg((unsigned)cdiv(N, 128), (unsigned)(c.F / 64));
    k_gemm_bf16_wmma<<<gg, 256, 0, stream>>>((const bf16_t*)xbf, (const bf16_t*)wt,
                                             hbuf, N, c.F, c.K);
    k_logits<<<cdiv((long)N * c.H, 256), 256, 0, stream>>>(
        hbuf, c.as, c.ad, es, ed, N, c.F, c.C, c.hShift, c.H - 1);
    k_fill<<<cdiv((long)N * c.H, 256), 256, 0, stream>>>(mbuf, -__builtin_inff(), N * c.H);
    k_fill<<<cdiv((long)N * c.H, 256), 256, 0, stream>>>(sbuf, 0.f, N * c.H);
    k_fill<<<cdiv((long)N * c.F, 256), 256, 0, stream>>>(agg, 0.f, N * c.F);
    k_edge_max<<<cdiv((long)Etot * c.H, 256), 256, 0, stream>>>(
        ei, es, ed, mbuf, E, Etot, c.hShift, c.H - 1);
    k_edge_sum<<<cdiv((long)Etot * c.H, 256), 256, 0, stream>>>(
        ei, es, ed, mbuf, sbuf, E, Etot, c.hShift, c.H - 1);
    k_edge_agg<<<cdiv(Etot, 8), 256, 0, stream>>>(
        ei, es, ed, mbuf, sbuf, hbuf, agg, E, Etot, c.F, c.H, c.cphShift);
    k_bn_stats<<<c.F, 256, 0, stream>>>(agg, N, c.F, mu, rs);
    if (li < 2) {
      k_bn_apply<<<cdiv((long)N * c.F, 256), 256, 0, stream>>>(
          agg, c.g, c.be, mu, rs, xbf, nullptr, N * c.F, c.F - 1, c.relu);
    } else {
      k_bn_apply<<<cdiv((long)N * c.F, 256), 256, 0, stream>>>(
          agg, c.g, c.be, mu, rs, nullptr, (float*)d_out, N * c.F, c.F - 1, c.relu);
    }
  }
}
